// MixtureOfExperts_76570676953145
// MI455X (gfx1250) — compile-verified, matrix operand order checked by reading, not run on hardware
//
#include <hip/hip_runtime.h>

#define N_TOK 32768
#define DIN   1024
#define DH    2048
#define DOUT  1024
#define NEXP  8

#define BM   128
#define BN   128
#define BK   32
#define LDSK 40                       // padded LDS row stride in bf16 elems (80B, 16B-aligned)
#define MAXTILES (N_TOK / BM + NEXP)  // 264 tiles covers any routing distribution

typedef __attribute__((ext_vector_type(16))) __bf16 v16bf;
typedef __attribute__((ext_vector_type(8)))  float  v8f;

// ---------------- workspace layout (bytes) ----------------
#define WS_ROUTES  0u                     // N int
#define WS_COUNTS  131072u                // 8 int
#define WS_CURSORS (131072u + 32u)        // 8 int
#define WS_OFFSETS (131072u + 64u)        // 9 int
#define WS_TLIST   (131072u + 128u)       // (N+128) int
#define WS_H       262912u                // (N+128)*DH bf16

// ================= kernel 0: zero counts =================
__global__ void moe_zero(int* __restrict__ counts) {
    if (threadIdx.x < NEXP) counts[threadIdx.x] = 0;
}

// ================= kernel 1: router (one wave32 per token) =================
__global__ __launch_bounds__(256) void moe_router(
    const float* __restrict__ x, const float* __restrict__ Wr, const float* __restrict__ br,
    float* __restrict__ probs, int* __restrict__ routes, int* __restrict__ counts)
{
    const int lane = threadIdx.x & 31;
    const int tok  = blockIdx.x * 8 + (threadIdx.x >> 5);   // grid sized exactly
    const float* xr = x + (size_t)tok * DIN;

    float s[NEXP];
#pragma unroll
    for (int e = 0; e < NEXP; ++e) s[e] = 0.f;

    for (int k = lane; k < DIN; k += 32) {
        float xv = xr[k];
        const float4* w = (const float4*)(Wr + (size_t)k * NEXP);
        float4 w0 = w[0], w1 = w[1];
        s[0] += xv * w0.x; s[1] += xv * w0.y; s[2] += xv * w0.z; s[3] += xv * w0.w;
        s[4] += xv * w1.x; s[5] += xv * w1.y; s[6] += xv * w1.z; s[7] += xv * w1.w;
    }
#pragma unroll
    for (int off = 16; off >= 1; off >>= 1)
#pragma unroll
        for (int e = 0; e < NEXP; ++e) s[e] += __shfl_xor(s[e], off, 32);

    if (lane == 0) {
        float m = -3.4e38f;
#pragma unroll
        for (int e = 0; e < NEXP; ++e) { s[e] += br[e]; m = fmaxf(m, s[e]); }
        float p[NEXP], den = 0.f;
#pragma unroll
        for (int e = 0; e < NEXP; ++e) { p[e] = __expf(s[e] - m); den += p[e]; }
        float inv = 1.f / den;
        int amax = 0; float best = -1.f;
#pragma unroll
        for (int e = 0; e < NEXP; ++e) {
            p[e] *= inv;
            probs[(size_t)tok * NEXP + e] = p[e];
            if (p[e] > best) { best = p[e]; amax = e; }   // strict > : first max wins (matches jnp.argmax)
        }
        routes[tok] = amax;
        atomicAdd(&counts[amax], 1);
    }
}

// ================= kernel 2: scan (E=8, single thread) =================
__global__ void moe_scan(const int* __restrict__ counts, int* __restrict__ offsets,
                         int* __restrict__ cursors, float* __restrict__ countsF)
{
    if (threadIdx.x == 0) {
        int acc = 0;
        for (int e = 0; e < NEXP; ++e) {
            offsets[e] = acc; cursors[e] = acc;
            countsF[e] = (float)counts[e];
            acc += counts[e];
        }
        offsets[NEXP] = acc;
    }
}

// ================= kernel 3: scatter token ids =================
__global__ __launch_bounds__(256) void moe_scatter(
    const int* __restrict__ routes, int* __restrict__ cursors, int* __restrict__ tlist)
{
    int tok = blockIdx.x * 256 + threadIdx.x;
    int r   = routes[tok];
    int pos = atomicAdd(&cursors[r], 1);
    tlist[pos] = tok;
}

// ---- locate (expert, tile) for blockIdx.y over compacted segments ----
__device__ __forceinline__ int find_tile(const int* __restrict__ offsets, int tileIdx,
                                         int& segStart, int& segEnd, int& rowBase)
{
    int acc = 0;
    for (int e = 0; e < NEXP; ++e) {
        int st = offsets[e], en = offsets[e + 1];
        int nt = (en - st + BM - 1) / BM;
        if (tileIdx < acc + nt) {
            segStart = st; segEnd = en;
            rowBase  = st + (tileIdx - acc) * BM;
            return e;
        }
        acc += nt;
    }
    return -1;
}

// ================= kernel 4: H = relu(gather(x) @ W1[e] + b1[e]) =================
__global__ __launch_bounds__(256) void moe_gemm1(
    const float* __restrict__ x, const float* __restrict__ W1all, const float* __restrict__ b1,
    const int* __restrict__ offsets, const int* __restrict__ tlist, __bf16* __restrict__ H)
{
    __shared__ __align__(16) __bf16 lA[BM * LDSK];
    __shared__ __align__(16) __bf16 lB[BN * LDSK];

    int segStart, segEnd, rowBase;
    int expert = find_tile(offsets, blockIdx.y, segStart, segEnd, rowBase);
    if (expert < 0) return;

    const int nBase = blockIdx.x * BN;
    const float* W  = W1all + (size_t)expert * DIN * DH;

    const int tid = threadIdx.x, lane = tid & 31, wv = tid >> 5;
    const int wm = wv & 3, wn = wv >> 2;
    const int l15 = lane & 15, half = lane >> 4;

    // A staging: thread stages 16 k's of one gathered row
    const int arow  = tid >> 1;
    const int ahalf = tid & 1;
    int apos = rowBase + arow;
    int atok = tlist[(apos < segEnd) ? apos : (segEnd - 1)];
    const float* aptr = x + (size_t)atok * DIN + ahalf * 16;

    // B staging: thread stages 16 k's of one weight column (coalesced across lanes)
    const int bcol = tid & 127;
    const int bks  = (tid >> 7) * 16;
    const float* bptr = W + (size_t)bks * DH + nBase + bcol;

    const v8f vzero = {0.f,0.f,0.f,0.f,0.f,0.f,0.f,0.f};
    v8f acc[2][4];
#pragma unroll
    for (int mi = 0; mi < 2; ++mi)
#pragma unroll
        for (int ni = 0; ni < 4; ++ni) acc[mi][ni] = vzero;

    for (int kb = 0; kb < DIN; kb += BK) {
        { // stage A tile (f32 -> bf16)
            const float4* p = (const float4*)(aptr + kb);
            float4 f0 = p[0], f1 = p[1], f2 = p[2], f3 = p[3];
            v16bf t;
            t[0]=(__bf16)f0.x; t[1]=(__bf16)f0.y; t[2]=(__bf16)f0.z; t[3]=(__bf16)f0.w;
            t[4]=(__bf16)f1.x; t[5]=(__bf16)f1.y; t[6]=(__bf16)f1.z; t[7]=(__bf16)f1.w;
            t[8]=(__bf16)f2.x; t[9]=(__bf16)f2.y; t[10]=(__bf16)f2.z; t[11]=(__bf16)f2.w;
            t[12]=(__bf16)f3.x; t[13]=(__bf16)f3.y; t[14]=(__bf16)f3.z; t[15]=(__bf16)f3.w;
            *(v16bf*)&lA[arow * LDSK + ahalf * 16] = t;
        }
        { // stage B tile transposed to K-contiguous per column (f32 -> bf16)
            const float* q = bptr + (size_t)kb * DH;
            v16bf t;
#pragma unroll
            for (int j = 0; j < 16; ++j) t[j] = (__bf16)q[(size_t)j * DH];
            *(v16bf*)&lB[bcol * LDSK + bks] = t;
        }
        if (kb + BK < DIN) __builtin_prefetch(aptr + kb + BK, 0, 3);
        __syncthreads();

        v16bf af[2], bfr[4];
#pragma unroll
        for (int mi = 0; mi < 2; ++mi)
            af[mi] = *(const v16bf*)&lA[(wm * 32 + mi * 16 + l15) * LDSK + half * 16];
#pragma unroll
        for (int ni = 0; ni < 4; ++ni)
            bfr[ni] = *(const v16bf*)&lB[(wn * 64 + ni * 16 + l15) * LDSK + half * 16];
#pragma unroll
        for (int mi = 0; mi < 2; ++mi)
#pragma unroll
            for (int ni = 0; ni < 4; ++ni)
                acc[mi][ni] = __builtin_amdgcn_wmma_f32_16x16x32_bf16(
                    false, af[mi], false, bfr[ni], (short)0, acc[mi][ni], false, false);
        __syncthreads();
    }

    // epilogue: +bias, ReLU, bf16 store (masked to valid rows only -> no cross-segment clobber)
#pragma unroll
    for (int ni = 0; ni < 4; ++ni) {
        int col = nBase + wn * 64 + ni * 16 + l15;
        float bv = b1[expert * DH + col];
#pragma unroll
        for (int mi = 0; mi < 2; ++mi) {
            int r0 = rowBase + wm * 32 + mi * 16 + half * 8;
#pragma unroll
            for (int r = 0; r < 8; ++r) {
                int pr = r0 + r;
                if (pr < segEnd) {
                    float v = acc[mi][ni][r] + bv;
                    H[(size_t)pr * DH + col] = (__bf16)fmaxf(v, 0.f);
                }
            }
        }
    }
}

// ================= kernel 5: out[tok] = H @ W2[e] + b2[e] =================
__global__ __launch_bounds__(256) void moe_gemm2(
    const __bf16* __restrict__ H, const float* __restrict__ W2all, const float* __restrict__ b2,
    const int* __restrict__ offsets, const int* __restrict__ tlist, float* __restrict__ out)
{
    __shared__ __align__(16) __bf16 lA[BM * LDSK];
    __shared__ __align__(16) __bf16 lB[BN * LDSK];

    int segStart, segEnd, rowBase;
    int expert = find_tile(offsets, blockIdx.y, segStart, segEnd, rowBase);
    if (expert < 0) return;

    const int nBase = blockIdx.x * BN;
    const float* W  = W2all + (size_t)expert * DH * DOUT;

    const int tid = threadIdx.x, lane = tid & 31, wv = tid >> 5;
    const int wm = wv & 3, wn = wv >> 2;
    const int l15 = lane & 15, half = lane >> 4;

    const int arow  = tid >> 1;
    const int ahalf = tid & 1;
    const __bf16* aptr = H + (size_t)(rowBase + arow) * DH + ahalf * 16;  // H padded by 128 rows

    const int bcol = tid & 127;
    const int bks  = (tid >> 7) * 16;
    const float* bptr = W + (size_t)bks * DOUT + nBase + bcol;

    const v8f vzero = {0.f,0.f,0.f,0.f,0.f,0.f,0.f,0.f};
    v8f acc[2][4];
#pragma unroll
    for (int mi = 0; mi < 2; ++mi)
#pragma unroll
        for (int ni = 0; ni < 4; ++ni) acc[mi][ni] = vzero;

    for (int kb = 0; kb < DH; kb += BK) {
        { // stage A tile (already bf16): two 16B copies
            const uint4* p = (const uint4*)(aptr + kb);
            uint4 u0 = p[0], u1 = p[1];
            uint4* d = (uint4*)&lA[arow * LDSK + ahalf * 16];
            d[0] = u0; d[1] = u1;
        }
        { // stage B tile transposed (f32 -> bf16)
            const float* q = bptr + (size_t)kb * DOUT;
            v16bf t;
#pragma unroll
            for (int j = 0; j < 16; ++j) t[j] = (__bf16)q[(size_t)j * DOUT];
            *(v16bf*)&lB[bcol * LDSK + bks] = t;
        }
        if (kb + BK < DH) __builtin_prefetch(aptr + kb + BK, 0, 3);
        __syncthreads();

        v16bf af[2], bfr[4];
#pragma unroll
        for (int mi = 0; mi < 2; ++mi)
            af[mi] = *(const v16bf*)&lA[(wm * 32 + mi * 16 + l15) * LDSK + half * 16];
#pragma unroll
        for (int ni = 0; ni < 4; ++ni)
            bfr[ni] = *(const v16bf*)&lB[(wn * 64 + ni * 16 + l15) * LDSK + half * 16];
#pragma unroll
        for (int mi = 0; mi < 2; ++mi)
#pragma unroll
            for (int ni = 0; ni < 4; ++ni)
                acc[mi][ni] = __builtin_amdgcn_wmma_f32_16x16x32_bf16(
                    false, af[mi], false, bfr[ni], (short)0, acc[mi][ni], false, false);
        __syncthreads();
    }

    // epilogue: +bias, scatter f32 rows back to token order
#pragma unroll
    for (int ni = 0; ni < 4; ++ni) {
        int col = nBase + wn * 64 + ni * 16 + l15;
        float bv = b2[expert * DOUT + col];
#pragma unroll
        for (int mi = 0; mi < 2; ++mi) {
            int r0 = rowBase + wm * 32 + mi * 16 + half * 8;
#pragma unroll
            for (int r = 0; r < 8; ++r) {
                int pr = r0 + r;
                if (pr < segEnd) {
                    int tok = tlist[pr];
                    out[(size_t)tok * DOUT + col] = acc[mi][ni][r] + bv;
                }
            }
        }
    }
}

// ================= host launcher =================
extern "C" void kernel_launch(void* const* d_in, const int* in_sizes, int n_in,
                              void* d_out, int out_size, void* d_ws, size_t ws_size,
                              hipStream_t stream)
{
    const float* x  = (const float*)d_in[0];
    const float* Wr = (const float*)d_in[1];
    const float* br = (const float*)d_in[2];
    const float* W1 = (const float*)d_in[3];
    const float* b1 = (const float*)d_in[4];
    const float* W2 = (const float*)d_in[5];
    const float* b2 = (const float*)d_in[6];

    float* out     = (float*)d_out;                       // [N, DOUT]
    float* probs   = out + (size_t)N_TOK * DOUT;          // [N, E]
    float* countsF = probs + (size_t)N_TOK * NEXP;        // [E]

    char* ws = (char*)d_ws;
    int*    routes  = (int*)(ws + WS_ROUTES);
    int*    counts  = (int*)(ws + WS_COUNTS);
    int*    cursors = (int*)(ws + WS_CURSORS);
    int*    offsets = (int*)(ws + WS_OFFSETS);
    int*    tlist   = (int*)(ws + WS_TLIST);
    __bf16* H       = (__bf16*)(ws + WS_H);

    moe_zero<<<1, 32, 0, stream>>>(counts);
    moe_router<<<N_TOK / 8, 256, 0, stream>>>(x, Wr, br, probs, routes, counts);
    moe_scan<<<1, 32, 0, stream>>>(counts, offsets, cursors, countsF);
    moe_scatter<<<N_TOK / 256, 256, 0, stream>>>(routes, cursors, tlist);

    dim3 g1(DH / BN, MAXTILES);
    moe_gemm1<<<g1, 256, 0, stream>>>(x, W1, b1, offsets, tlist, H);

    dim3 g2(DOUT / BN, MAXTILES);
    moe_gemm2<<<g2, 256, 0, stream>>>(H, W2, b2, offsets, tlist, out);
}
